// ComparisonLayer_55877524521600
// MI455X (gfx1250) — matrix operand order repaired and checked by reading, not compiled, not run on hardware
//
#include <hip/hip_runtime.h>
#include <hip/hip_bf16.h>

typedef __attribute__((ext_vector_type(16))) _Float16 v16h;
typedef __attribute__((ext_vector_type(8)))  _Float16 v8h;
typedef __attribute__((ext_vector_type(8)))  float    v8f;

#define L_DIM 512
#define B_DIM 2
#define D_DIM 256
#define C_DIM 128
#define O_DIM 64
#define BT_STRIDE 136   // 64 o-rows x 136 halfs: 272B row stride -> conflict-free, 16B aligned

// ---------------- Phase 1: LN + a/b/a4/b4 projections (small) ----------------
__global__ __launch_bounds__(256) void cmp_phase1(
    const float* __restrict__ x, const float* __restrict__ g, const float* __restrict__ be,
    const float* __restrict__ w1, const float* __restrict__ b1,
    const float* __restrict__ w2, const float* __restrict__ b2,
    const float* __restrict__ w4,
    _Float16* __restrict__ a_h, float* __restrict__ b_f,
    float* __restrict__ a4, float* __restrict__ b4)
{
  __shared__ float red[D_DIM];
  __shared__ float xln[D_DIM];
  __shared__ float sa[C_DIM];
  __shared__ float sb[C_DIM];
  const int rid = blockIdx.x;          // rid = bb*L + l  (row of (B,L) space)
  const int bb  = rid / L_DIM;
  const int l   = rid % L_DIM;
  const int tid = threadIdx.x;         // 256 threads == D

  // x is (L, B, D)
  const float xd = x[((size_t)l * B_DIM + bb) * D_DIM + tid];

  red[tid] = xd; __syncthreads();
  for (int s = 128; s > 0; s >>= 1) { if (tid < s) red[tid] += red[tid + s]; __syncthreads(); }
  const float mu = red[0] * (1.0f / D_DIM);
  __syncthreads();
  const float dv = xd - mu;
  red[tid] = dv * dv; __syncthreads();
  for (int s = 128; s > 0; s >>= 1) { if (tid < s) red[tid] += red[tid + s]; __syncthreads(); }
  const float var = red[0] * (1.0f / D_DIM);
  xln[tid] = dv * rsqrtf(var + 1e-5f) * g[tid] + be[tid];
  __syncthreads();

  if (tid < C_DIM) {                    // a = xln @ w1 + b1
    const int c = tid;
    float acc = b1[c];
    #pragma unroll 4
    for (int d = 0; d < D_DIM; ++d) acc = fmaf(xln[d], w1[d * C_DIM + c], acc);
    sa[c] = acc;
    a_h[(size_t)rid * C_DIM + c] = (_Float16)acc;
  } else {                              // b = xln @ w2 + b2
    const int c = tid - C_DIM;
    float acc = b2[c];
    #pragma unroll 4
    for (int d = 0; d < D_DIM; ++d) acc = fmaf(xln[d], w2[d * C_DIM + c], acc);
    sb[c] = acc;
    b_f[(size_t)rid * C_DIM + c] = acc;
  }
  __syncthreads();

  if (tid < O_DIM) {                    // a4 = a @ w4
    const int o = tid;
    float acc = 0.0f;
    #pragma unroll 4
    for (int c = 0; c < C_DIM; ++c) acc = fmaf(sa[c], w4[c * O_DIM + o], acc);
    a4[(size_t)rid * O_DIM + o] = acc;
  } else if (tid < 2 * O_DIM) {         // b4 = b @ w4
    const int o = tid - O_DIM;
    float acc = 0.0f;
    #pragma unroll 4
    for (int c = 0; c < C_DIM; ++c) acc = fmaf(sb[c], w4[c * O_DIM + o], acc);
    b4[(size_t)rid * O_DIM + o] = acc;
  }
}

// ------------- Phase 2: per (bb,m): out_m[l,o] = a[l,:] @ (b[m,:]*w3) --------
__global__ __launch_bounds__(256) void cmp_phase2(
    const _Float16* __restrict__ a_h, const float* __restrict__ b_f,
    const float* __restrict__ w3, const float* __restrict__ b3,
    const float* __restrict__ a4, const float* __restrict__ b4,
    float* __restrict__ out)
{
  __shared__ __attribute__((aligned(32))) _Float16 bt[O_DIM * BT_STRIDE]; // B_m^T [o][c], f16
  __shared__ float sbm[C_DIM];

  const int wg  = blockIdx.x;
  const int lb  = wg & 3;                 // which 128-row l block
  const int m   = (wg >> 2) & (L_DIM - 1);
  const int bb  = wg >> 11;
  const int tid = threadIdx.x;

  if (tid < C_DIM) sbm[tid] = b_f[((size_t)(bb * L_DIM + m)) * C_DIM + tid];
  __syncthreads();

  // Build B_m^T in LDS: bt[o][c] = f16(b[m,c] * w3[c,o]); coalesced w3 reads over o.
  {
    const int o  = tid & (O_DIM - 1);
    const int cb = (tid >> 6) << 5;       // 4 thread groups x 32 c each
    #pragma unroll
    for (int cc = 0; cc < 32; ++cc) {
      const int c = cb + cc;
      bt[o * BT_STRIDE + c] = (_Float16)(sbm[c] * w3[c * O_DIM + o]);
    }
  }
  __syncthreads();

  const int lane = tid & 31;
  const int wave = tid >> 5;
  const int row  = lane & 15;             // M-row (A) / N-col (B,C,D)
  const int half = lane >> 4;             // lane-half selects K sub-range
  const int ltile = lb * 128 + wave * 16;

  // A fragments: 16x32 f16, ISA layout: j<8 -> K=c0+8*half+j ; j>=8 -> K=c0+16+8*half+(j-8)
  v16h afrag[4];
  const _Float16* ap = a_h + ((size_t)(bb * L_DIM + ltile + row)) * C_DIM;
  #pragma unroll
  for (int kk = 0; kk < 4; ++kk) {
    const int c0 = kk * 32;
    const v8h lo = *(const v8h*)(ap + c0 + 8 * half);
    const v8h hi = *(const v8h*)(ap + c0 + 16 + 8 * half);
    #pragma unroll
    for (int j = 0; j < 8; ++j) { afrag[kk][j] = lo[j]; afrag[kk][j + 8] = hi[j]; }
  }

  const size_t mrow = (size_t)(bb * L_DIM + m);

  #pragma unroll
  for (int ot = 0; ot < 4; ++ot) {
    const int o0 = ot * 16;
    // lane-constant part of the result: +b3[o] - b4[m,o]
    const float cinit = b3[o0 + row] - b4[mrow * O_DIM + o0 + row];
    v8f acc;
    #pragma unroll
    for (int r = 0; r < 8; ++r) acc[r] = cinit;

    #pragma unroll
    for (int kk = 0; kk < 4; ++kk) {
      // B fragment 32x16 f16: N = lane&15, j<8 -> K = kk*32 + 8*half + j ; j>=8 -> K = kk*32 + 16 + 8*half + (j-8)
      const _Float16* bp = &bt[(o0 + row) * BT_STRIDE + kk * 32 + 8 * half];
      const v8h lo = *(const v8h*)bp;
      const v8h hi = *(const v8h*)(bp + 16);
      v16h bfrag;
      #pragma unroll
      for (int j = 0; j < 8; ++j) { bfrag[j] = lo[j]; bfrag[j + 8] = hi[j]; }
      acc = __builtin_amdgcn_wmma_f32_16x16x32_f16(
          /*neg_a=*/false, afrag[kk], /*neg_b=*/false, bfrag,
          /*c_mod=*/(short)0, acc, /*reuse_a=*/false, /*reuse_b=*/false);
    }

    // D layout: VGPR r -> l-row = r + 8*half ; col = lane&15. Add per-row a4[l,o]
    // and stream the 128MB result past L2 with non-temporal stores (no reuse;
    // keeps the 192MB L2 clean for a_h/b_f/a4/b4/w3 which ARE re-read per m).
    #pragma unroll
    for (int r = 0; r < 8; ++r) {
      const int l_r = ltile + r + 8 * half;
      const size_t lidx = (size_t)(bb * L_DIM + l_r);
      const size_t idx  = ((lidx * L_DIM) + m) * O_DIM + o0 + row;
      __builtin_nontemporal_store(acc[r] + a4[lidx * O_DIM + o0 + row], &out[idx]);
    }
  }
}

extern "C" void kernel_launch(void* const* d_in, const int* in_sizes, int n_in,
                              void* d_out, int out_size, void* d_ws, size_t ws_size,
                              hipStream_t stream) {
  (void)in_sizes; (void)n_in; (void)out_size; (void)ws_size;
  const float* x  = (const float*)d_in[0];
  const float* g  = (const float*)d_in[1];
  const float* be = (const float*)d_in[2];
  const float* w1 = (const float*)d_in[3];
  const float* b1 = (const float*)d_in[4];
  const float* w2 = (const float*)d_in[5];
  const float* b2 = (const float*)d_in[6];
  const float* w3 = (const float*)d_in[7];
  const float* b3 = (const float*)d_in[8];
  const float* w4 = (const float*)d_in[9];
  float* out = (float*)d_out;

  // Workspace layout (bytes): a_h f16 [1024*128], b_f f32 [1024*128],
  //                           a4 f32 [1024*64], b4 f32 [1024*64]  -> 1.25 MB total
  char* ws = (char*)d_ws;
  _Float16* a_h = (_Float16*)(ws);
  float*    b_f = (float*)(ws + 262144);
  float*    a4  = (float*)(ws + 786432);
  float*    b4  = (float*)(ws + 1048576);

  cmp_phase1<<<B_DIM * L_DIM, 256, 0, stream>>>(x, g, be, w1, b1, w2, b2, w4,
                                                a_h, b_f, a4, b4);
  // 4096 WGs: bb(2) * m(512) * l-block(4); 8 waves each -> 16 WMMAs/wave
  cmp_phase2<<<B_DIM * L_DIM * 4, 256, 0, stream>>>(a_h, b_f, w3, b3, a4, b4, out);
}
